// Decoder_59450937312001
// MI455X (gfx1250) — matrix operand order repaired and checked by reading, not compiled, not run on hardware
//
#include <hip/hip_runtime.h>
#include <stdint.h>

// ---------------------------------------------------------------------------
// Types for CDNA5 WMMA (wave32, 16x16x32 bf16 -> f32) and TDM descriptors
// ---------------------------------------------------------------------------
typedef __attribute__((ext_vector_type(8)))  __bf16 v8bf;
typedef __attribute__((ext_vector_type(16))) __bf16 v16bf;
typedef __attribute__((ext_vector_type(8)))  float  v8f;
typedef __attribute__((ext_vector_type(4)))  unsigned int u32x4;
typedef __attribute__((ext_vector_type(8)))  int i32x8;
typedef __attribute__((ext_vector_type(4)))  int i32x4;

#define NBLK 64          // workgroups in the persistent LSTM kernel
#define TSTEPS 1000
#define BATCH 32

#if __has_builtin(__builtin_amdgcn_tensor_load_to_lds)
#define HAVE_TDM 1
#else
#define HAVE_TDM 0
#endif

// f32 -> bf16 round-to-nearest-even, stored as raw u16 bits
__device__ __forceinline__ unsigned short f2bf(float x) {
  unsigned u = __float_as_uint(x);
  unsigned r = 0x7FFFu + ((u >> 16) & 1u);
  return (unsigned short)((u + r) >> 16);
}

__device__ __forceinline__ float sigf(float x) {
  return 1.0f / (1.0f + __expf(-x));
}

// A-matrix fragment (16x32 bf16): lane L (L<16) row M=L holds K[kOff..kOff+7]
// and K[kOff+16..kOff+23], kOff = (lane>=16)*8.  Two 16B loads.
__device__ __forceinline__ v16bf load_fragA(const unsigned short* row, int k, int lane) {
  const __bf16* p = (const __bf16*)(row + k + ((lane >> 4) << 3));
  v8bf lo = *(const v8bf*)p;
  v8bf hi = *(const v8bf*)(p + 16);
  v16bf v;
#pragma unroll
  for (int i = 0; i < 8; ++i) { v[i] = lo[i]; v[8 + i] = hi[i]; }
  return v;
}

// B-matrix fragment (32x16 bf16): lane L holds column N=L&15, K[kOff..kOff+7]
// and K[kOff+16..kOff+23], kOff = (lane>=16)*8 (W stored [N,K] row-major).
__device__ __forceinline__ v16bf load_fragB(const unsigned short* row, int k, int lane) {
  const __bf16* p = (const __bf16*)(row + k + ((lane >> 4) << 3));
  v8bf lo = *(const v8bf*)p;
  v8bf hi = *(const v8bf*)(p + 16);
  v16bf v;
#pragma unroll
  for (int i = 0; i < 8; ++i) { v[i] = lo[i]; v[8 + i] = hi[i]; }
  return v;
}

__device__ __forceinline__ v8f wmma_bf16(v16bf a, v16bf b, v8f c) {
  return __builtin_amdgcn_wmma_f32_16x16x32_bf16(false, a, false, b, (short)0, c,
                                                 false, false);
}

// Grid-wide barrier: monotonically increasing counter, release on arrive,
// acquire on spin (agent scope -> proper L0/L2 ordering of h-state).
__device__ __forceinline__ void grid_barrier(unsigned* cnt, unsigned target) {
  __syncthreads();
  if (threadIdx.x == 0) {
    __hip_atomic_fetch_add(cnt, 1u, __ATOMIC_RELEASE, __HIP_MEMORY_SCOPE_AGENT);
    while (__hip_atomic_load(cnt, __ATOMIC_ACQUIRE, __HIP_MEMORY_SCOPE_AGENT) < target) {
      __builtin_amdgcn_s_sleep(1);
    }
  }
  __syncthreads();
}

#if HAVE_TDM
// Issue one 2D TDM load: `rows` consecutive rows of `cols` bf16 elements from
// a row-major global matrix into LDS at byte offset lds_off (rows packed
// contiguously).  D# bit layout per CDNA5 ISA ch.8 (groups 0/1; 2D -> groups
// 2/3 zero).  data_size=1 (2 bytes/elem), type=2 ("image"), count=1.
// clang-23 toolchain: 6-arg builtin (g0 u32x4, g1 i32x8, g2 i32x4, g3 i32x4,
// extra i32x8, i32 cpol).
__device__ __forceinline__ void tdm_load_2d(const void* gptr, unsigned cols,
                                            unsigned rows, unsigned lds_off) {
  unsigned long long ga = (unsigned long long)(uintptr_t)gptr;
  u32x4 g0;
  g0[0] = 1u;                                     // count=1, user descriptor
  g0[1] = lds_off;                                // lds_addr (bytes)
  g0[2] = (unsigned)(ga & 0xFFFFFFFFu);           // global_addr[31:0]
  g0[3] = (unsigned)((ga >> 32) & 0x01FFFFFFu)    // global_addr[56:32]
          | (2u << 30);                           // type=2
  i32x8 g1;
  g1[0] = (int)(1u << 16);                        // data_size=1 (2B)
  g1[1] = (int)((cols & 0xFFFFu) << 16);          // tensor_dim0[15:0]
  g1[2] = (int)(((cols >> 16) & 0xFFFFu)          // tensor_dim0[31:16]
          | ((rows & 0xFFFFu) << 16));            // tensor_dim1[15:0]
  g1[3] = (int)((cols & 0xFFFFu) << 16);          // tensor_dim1[31:16]=0 | tile_dim0
  g1[4] = (int)(rows & 0xFFFFu);                  // tile_dim1 | tile_dim2=0
  g1[5] = (int)cols;                              // tensor_dim0_stride[31:0]
  g1[6] = 0;                                      // stride hi | dim1_stride lo
  g1[7] = 0;
  i32x4 z4 = {0, 0, 0, 0};
  i32x8 z8 = {0, 0, 0, 0, 0, 0, 0, 0};
  __builtin_amdgcn_tensor_load_to_lds(g0, g1, z4, z4, z8, 0);
}
#endif

// ---------------------------------------------------------------------------
// Setup / conversion kernels
// ---------------------------------------------------------------------------
__global__ void zero_u32(unsigned* p, int n) {
  int i = blockIdx.x * blockDim.x + threadIdx.x;
  if (i < n) p[i] = 0u;
}

__global__ void cvt_bf16_flat(const float* __restrict__ in,
                              unsigned short* __restrict__ out, long long n) {
  long long i = (long long)blockIdx.x * blockDim.x + threadIdx.x;
  if (i < n) out[i] = f2bf(in[i]);
}

// W1 [256,80] f32 -> [256,96] bf16 zero-padded (K padded to a multiple of 32)
__global__ void cvt_W1_pad(const float* __restrict__ W1,
                           unsigned short* __restrict__ out) {
  int i = blockIdx.x * blockDim.x + threadIdx.x;
  if (i >= 256 * 96) return;
  int r = i / 96, c = i - r * 96;
  out[i] = (c < 80) ? f2bf(W1[r * 80 + c]) : (unsigned short)0;
}

// prev[t*32+b][0..95] = (t>0 && c<80) ? y_mels[b][t-1][c] : 0   (time-major rows)
__global__ void build_prev(const float* __restrict__ y,
                           unsigned short* __restrict__ out) {
  long long i = (long long)blockIdx.x * blockDim.x + threadIdx.x;
  if (i >= (long long)BATCH * TSTEPS * 96) return;
  long long rr = i / 96;
  int c = (int)(i - rr * 96);
  int t = (int)(rr >> 5), b = (int)(rr & 31);
  float v = (c < 80 && t > 0)
                ? y[(long long)b * TSTEPS * 80 + (long long)(t - 1) * 80 + c]
                : 0.0f;
  out[i] = f2bf(v);
}

__global__ void add_bias(const float* __restrict__ a, const float* __restrict__ b,
                         float* __restrict__ o, int n) {
  int i = blockIdx.x * blockDim.x + threadIdx.x;
  if (i < n) o[i] = a[i] + b[i];
}

// ---------------------------------------------------------------------------
// Generic GEMM: C[M,N](bf16, ReLU) = A[M,K](bf16) * W[N,K]^T (bf16), f32 acc.
// One 16x16 output tile per wave; 8 waves / block.
// ---------------------------------------------------------------------------
__global__ void gemm_relu_bf16(const unsigned short* __restrict__ A, int lda,
                               const unsigned short* __restrict__ W, int ldw,
                               unsigned short* __restrict__ C, int ldc,
                               int tilesM, int tilesN, int K) {
  int wave = threadIdx.x >> 5, lane = threadIdx.x & 31;
  long long tile = (long long)blockIdx.x * 8 + wave;
  if (tile >= (long long)tilesM * tilesN) return;
  int tn = (int)(tile % tilesN);
  int tm = (int)(tile / tilesN);
  int lmod = lane & 15;
  const unsigned short* aRow = A + (long long)(tm * 16 + lmod) * lda;
  const unsigned short* wRow = W + (long long)(tn * 16 + lmod) * ldw;
  v8f acc = {};
  for (int k = 0; k < K; k += 32)
    acc = wmma_bf16(load_fragA(aRow, k, lane), load_fragB(wRow, k, lane), acc);
  int mh = (lane >> 4) << 3;
#pragma unroll
  for (int r = 0; r < 8; ++r) {
    int m = tm * 16 + r + mh;
    float v = fmaxf(acc[r], 0.0f);
    C[(long long)m * ldc + tn * 16 + lmod] = f2bf(v);
  }
}

// ---------------------------------------------------------------------------
// Persistent LSTM layer, 64 WGs x 8 waves, one grid barrier per step.
// WG g owns hidden columns [16g,16g+16) for all 4 gates.  Weight slices
// staged into LDS via the Tensor Data Mover (8 x 2D tensor_load_to_lds,
// drained with s_wait_tensorcnt); c-state lives in LDS; h double-buffered in
// global (all WGs read full h each step).
// LDS layout: sWih[64][Kx] | sWhh[64][1024] | sGate[4][32][16] | sC[32][16]
// xs0/xs1: x-input segments, addr = base + b*sB + t*sT (elements).
// ---------------------------------------------------------------------------
__global__ void lstm_rec(const unsigned short* __restrict__ xs0, int k0,
                         long long s0B, long long s0T,
                         const unsigned short* __restrict__ xs1, int k1,
                         long long s1B, long long s1T,
                         const unsigned short* __restrict__ wih,
                         const unsigned short* __restrict__ whh,
                         const float* __restrict__ bias,
                         unsigned short* __restrict__ h_db,   // [2][32][1024]
                         unsigned short* __restrict__ h_seq,  // [T][32][1024]
                         unsigned* sync, int T) {
  extern __shared__ char smemRaw[];
  const int Kx = k0 + k1;
  const unsigned wihBytes = (unsigned)(64u * (unsigned)Kx * 2u);
  unsigned short* sWih = (unsigned short*)smemRaw;                 // [64][Kx]
  unsigned short* sWhh = (unsigned short*)(smemRaw + wihBytes);    // [64][1024]
  float* sGate = (float*)(smemRaw + wihBytes + 64 * 1024 * 2);     // [4][32][16]
  float* sC = sGate + 4 * 32 * 16;                                 // [32][16]

  const int tid = threadIdx.x;
  const int wg = blockIdx.x;
  const int wave = tid >> 5, lane = tid & 31;
  const int mi = wave & 1;        // batch half: rows mi*16..mi*16+15
  const int gi = wave >> 1;       // gate 0..3 (i,f,g,o)
  const int lmod = lane & 15;

  // ---- Stage this WG's weight slice (64 gate rows) into LDS ----
#if HAVE_TDM
  if (wave == 0) {
    // One 16-row x K-col tile per gate group per weight matrix.
#pragma unroll
    for (int g = 0; g < 4; ++g) {
      int nbase = g * 1024 + wg * 16;
      tdm_load_2d(wih + (long long)nbase * Kx, (unsigned)Kx, 16u,
                  (unsigned)(g * 16) * (unsigned)Kx * 2u);
      tdm_load_2d(whh + (long long)nbase * 1024, 1024u, 16u,
                  wihBytes + (unsigned)(g * 16) * 1024u * 2u);
    }
    __builtin_amdgcn_s_wait_tensorcnt(0);
  }
#else
  {
    const int kx8 = Kx >> 3;
    for (int i = tid; i < 64 * kx8; i += 256) {
      int lr = i / kx8;
      int kc = (i - lr * kx8) << 3;
      int n = (lr >> 4) * 1024 + wg * 16 + (lr & 15);
      *(v8bf*)(sWih + (long long)lr * Kx + kc) =
          *(const v8bf*)(wih + (long long)n * Kx + kc);
    }
    for (int i = tid; i < 64 * 128; i += 256) {
      int lr = i >> 7;
      int kc = (i & 127) << 3;
      int n = (lr >> 4) * 1024 + wg * 16 + (lr & 15);
      *(v8bf*)(sWhh + (long long)lr * 1024 + kc) =
          *(const v8bf*)(whh + (long long)n * 1024 + kc);
    }
  }
#endif
  // c = 0, h[buffer 0, our slice] = 0
  for (int e = tid; e < 512; e += 256) {
    sC[e] = 0.0f;
    int b = e >> 4, j = e & 15;
    h_db[b * 1024 + wg * 16 + j] = 0;
  }
  __syncthreads();
  unsigned tgt = 0;
  grid_barrier(sync, tgt += NBLK);

  const unsigned short* wBih = sWih + (long long)(gi * 16 + lmod) * Kx;
  const unsigned short* wBhh = sWhh + (long long)(gi * 16 + lmod) * 1024;
  const int arow = mi * 16 + lmod;

  for (int t = 0; t < T; ++t) {
    v8f acc = {};
    // x segment 0
    {
      const unsigned short* a0 = xs0 + (long long)arow * s0B + (long long)t * s0T;
      for (int k = 0; k < k0; k += 32)
        acc = wmma_bf16(load_fragA(a0, k, lane), load_fragB(wBih, k, lane), acc);
      if (t + 1 < T) __builtin_prefetch((const void*)(a0 + s0T), 0, 1);
    }
    // x segment 1 (layer 0: memory part)
    if (k1 > 0) {
      const unsigned short* a1 = xs1 + (long long)arow * s1B + (long long)t * s1T;
      for (int k = 0; k < k1; k += 32)
        acc = wmma_bf16(load_fragA(a1, k, lane), load_fragB(wBih + k0, k, lane), acc);
    }
    // recurrent segment: h_{t-1} from double buffer (t&1)
    {
      const unsigned short* ah = h_db + (t & 1) * (BATCH * 1024) + arow * 1024;
      for (int k = 0; k < 1024; k += 32)
        acc = wmma_bf16(load_fragA(ah, k, lane), load_fragB(wBhh, k, lane), acc);
    }
    // stash this wave's 16x16 gate tile into LDS (f32 C/D layout)
    {
      float* g = sGate + gi * 512;
#pragma unroll
      for (int r = 0; r < 8; ++r) {
        int m = mi * 16 + r + ((lane >> 4) << 3);
        g[m * 16 + lmod] = acc[r];
      }
    }
    __syncthreads();
    // elementwise cell update for our 32x16 slice (all four gates local)
    unsigned short* hw = h_db + ((t + 1) & 1) * (BATCH * 1024);
    for (int e = tid; e < 512; e += 256) {
      int b = e >> 4, j = e & 15;
      int jg = wg * 16 + j;
      float gI = sGate[0 * 512 + b * 16 + j] + bias[jg];
      float gF = sGate[1 * 512 + b * 16 + j] + bias[1024 + jg];
      float gG = sGate[2 * 512 + b * 16 + j] + bias[2048 + jg];
      float gO = sGate[3 * 512 + b * 16 + j] + bias[3072 + jg];
      float c = sC[e];
      c = sigf(gF) * c + sigf(gI) * tanhf(gG);
      float h = sigf(gO) * tanhf(c);
      sC[e] = c;
      unsigned short hb = f2bf(h);
      hw[b * 1024 + jg] = hb;
      h_seq[((long long)t * BATCH + b) * 1024 + jg] = hb;
    }
    grid_barrier(sync, tgt += NBLK);  // publish h_t to all WGs
  }
}

// ---------------------------------------------------------------------------
// Final projection: out[b*T+t][0..79] = [h1(t,b) | memory(b,t)] @ Wp^T + bias.
// Virtual concat: two K segments with different row mappings. f32 output.
// ---------------------------------------------------------------------------
__global__ void proj_gemm(const unsigned short* __restrict__ h1seq,
                          const unsigned short* __restrict__ memB,
                          const unsigned short* __restrict__ Wp,
                          const float* __restrict__ bproj,
                          float* __restrict__ out, int T) {
  int wave = threadIdx.x >> 5, lane = threadIdx.x & 31;
  int tilesM = (BATCH * T) >> 4;  // 2000
  long long tile = (long long)blockIdx.x * 8 + wave;
  if (tile >= (long long)tilesM * 5) return;
  int tn = (int)(tile % 5), tm = (int)(tile / 5);
  int lmod = lane & 15;
  int rm = tm * 16 + lmod;        // output row = b*T + t
  int b = rm / T, t = rm - b * T;
  const unsigned short* aH = h1seq + ((long long)t * BATCH + b) * 1024;
  const unsigned short* aM = memB + ((long long)b * T + t) * 512;
  const unsigned short* wRow = Wp + (long long)(tn * 16 + lmod) * 1536;
  v8f acc = {};
  for (int k = 0; k < 1024; k += 32)
    acc = wmma_bf16(load_fragA(aH, k, lane), load_fragB(wRow, k, lane), acc);
  for (int k = 0; k < 512; k += 32)
    acc = wmma_bf16(load_fragA(aM, k, lane), load_fragB(wRow + 1024, k, lane), acc);
  int mh = (lane >> 4) << 3;
#pragma unroll
  for (int r = 0; r < 8; ++r) {
    int m = tm * 16 + r + mh;
    out[(long long)m * 80 + tn * 16 + lmod] = acc[r] + bproj[tn * 16 + lmod];
  }
}

// ---------------------------------------------------------------------------
// Host launch
// ---------------------------------------------------------------------------
extern "C" void kernel_launch(void* const* d_in, const int* in_sizes, int n_in,
                              void* d_out, int out_size, void* d_ws, size_t ws_size,
                              hipStream_t stream) {
  (void)in_sizes; (void)n_in; (void)out_size; (void)ws_size;
  const int T = TSTEPS;
  const float* memory = (const float*)d_in[0];   // [32,1000,512]
  const float* y_mels = (const float*)d_in[1];   // [32,1000,80]
  const float* W1     = (const float*)d_in[2];   // [256,80]
  const float* W2     = (const float*)d_in[3];   // [256,256]
  const float* w_ih0  = (const float*)d_in[4];   // [4096,768]
  const float* w_hh0  = (const float*)d_in[5];   // [4096,1024]
  const float* b_ih0  = (const float*)d_in[6];
  const float* b_hh0  = (const float*)d_in[7];
  const float* w_ih1  = (const float*)d_in[8];   // [4096,1024]
  const float* w_hh1  = (const float*)d_in[9];   // [4096,1024]
  const float* b_ih1  = (const float*)d_in[10];
  const float* b_hh1  = (const float*)d_in[11];
  const float* W_proj = (const float*)d_in[12];  // [80,1536]
  const float* b_proj = (const float*)d_in[13];

  char* ws = (char*)d_ws;
  size_t off = 0;
  auto alloc = [&](size_t bytes) -> void* {
    void* p = ws + off;
    off = (off + bytes + 255) & ~(size_t)255;
    return p;
  };
  unsigned short* mem_bf = (unsigned short*)alloc((size_t)32 * T * 512 * 2);
  unsigned short* wih0b  = (unsigned short*)alloc((size_t)4096 * 768 * 2);
  unsigned short* whh0b  = (unsigned short*)alloc((size_t)4096 * 1024 * 2);
  unsigned short* wih1b  = (unsigned short*)alloc((size_t)4096 * 1024 * 2);
  unsigned short* whh1b  = (unsigned short*)alloc((size_t)4096 * 1024 * 2);
  unsigned short* W1b    = (unsigned short*)alloc((size_t)256 * 96 * 2);
  unsigned short* W2b    = (unsigned short*)alloc((size_t)256 * 256 * 2);
  unsigned short* wpb    = (unsigned short*)alloc((size_t)80 * 1536 * 2);
  float*          bias0  = (float*)alloc((size_t)4096 * 4);
  float*          bias1  = (float*)alloc((size_t)4096 * 4);
  unsigned short* prevp  = (unsigned short*)alloc((size_t)32 * T * 96 * 2);
  unsigned short* p1     = (unsigned short*)alloc((size_t)32 * T * 256 * 2);
  unsigned short* p2     = (unsigned short*)alloc((size_t)32 * T * 256 * 2);
  unsigned short* h0seq  = (unsigned short*)alloc((size_t)32 * T * 1024 * 2);
  unsigned short* h1seq  = (unsigned short*)alloc((size_t)32 * T * 1024 * 2);
  unsigned short* hdb    = (unsigned short*)alloc((size_t)2 * 32 * 1024 * 2);
  unsigned*       sync   = (unsigned*)alloc(256);

  auto blocks = [](long long n) { return (unsigned)((n + 255) / 256); };

  zero_u32<<<1, 64, 0, stream>>>(sync, 8);
  cvt_bf16_flat<<<blocks((long long)32 * T * 512), 256, 0, stream>>>(memory, mem_bf, (long long)32 * T * 512);
  cvt_bf16_flat<<<blocks((long long)4096 * 768), 256, 0, stream>>>(w_ih0, wih0b, (long long)4096 * 768);
  cvt_bf16_flat<<<blocks((long long)4096 * 1024), 256, 0, stream>>>(w_hh0, whh0b, (long long)4096 * 1024);
  cvt_bf16_flat<<<blocks((long long)4096 * 1024), 256, 0, stream>>>(w_ih1, wih1b, (long long)4096 * 1024);
  cvt_bf16_flat<<<blocks((long long)4096 * 1024), 256, 0, stream>>>(w_hh1, whh1b, (long long)4096 * 1024);
  cvt_bf16_flat<<<blocks(256 * 256), 256, 0, stream>>>(W2, W2b, 256 * 256);
  cvt_bf16_flat<<<blocks(80 * 1536), 256, 0, stream>>>(W_proj, wpb, 80 * 1536);
  cvt_W1_pad<<<blocks(256 * 96), 256, 0, stream>>>(W1, W1b);
  build_prev<<<blocks((long long)32 * T * 96), 256, 0, stream>>>(y_mels, prevp);
  add_bias<<<16, 256, 0, stream>>>(b_ih0, b_hh0, bias0, 4096);
  add_bias<<<16, 256, 0, stream>>>(b_ih1, b_hh1, bias1, 4096);

  // Prenet: [32000,96]x[256,96]^T -> ReLU -> [32000,256]x[256,256]^T -> ReLU
  gemm_relu_bf16<<<4000, 256, 0, stream>>>(prevp, 96, W1b, 96, p1, 256, 2000, 16, 96);
  gemm_relu_bf16<<<4000, 256, 0, stream>>>(p1, 256, W2b, 256, p2, 256, 2000, 16, 256);

  // LSTM layers: weights TDM-staged into LDS (229KB / 262KB per WGP)
  size_t sh0 = (size_t)64 * (768 + 1024 + 1024) * 2 + (4 * 32 * 16 + 32 * 16) * 4;
  size_t sh1 = (size_t)64 * (1024 + 1024 + 1024) * 2 + (4 * 32 * 16 + 32 * 16) * 4;
  (void)hipFuncSetAttribute((const void*)lstm_rec,
                            hipFuncAttributeMaxDynamicSharedMemorySize, (int)sh1);
  lstm_rec<<<NBLK, 256, sh0, stream>>>(
      p2, 256, 256LL, 32LL * 256,                  // x seg0: prenet (time-major)
      mem_bf, 512, (long long)T * 512, 512LL,      // x seg1: memory [B,T,512]
      wih0b, whh0b, bias0, hdb, h0seq, sync + 0, T);
  lstm_rec<<<NBLK, 256, sh1, stream>>>(
      h0seq, 1024, 1024LL, 32LL * 1024,            // x seg0: layer-0 output
      (const unsigned short*)nullptr, 0, 0LL, 0LL,
      wih1b, whh1b, bias1, hdb, h1seq, sync + 1, T);

  // Projection: [h1 | memory] @ W_proj^T + b_proj -> d_out [32,1000,80] f32
  proj_gemm<<<1250, 256, 0, stream>>>(h1seq, mem_bf, wpb, b_proj, (float*)d_out, T);
}
